// SemanticSemanticAggregator_8718783611564
// MI455X (gfx1250) — matrix operand order repaired and checked by reading, not compiled, not run on hardware
//
#include <hip/hip_runtime.h>
#include <hip/hip_bf16.h>

// ---------------------------------------------------------------------------
// Types for CDNA5 WMMA
// ---------------------------------------------------------------------------
typedef __attribute__((ext_vector_type(16))) __bf16 v16bf;
typedef __attribute__((ext_vector_type(8)))  float  v8f;
typedef int v4i __attribute__((vector_size(16)));   // matches builtin's int4

#define BM 128
#define BN 128
#define BK 32

// CDNA5 async global->LDS copy path (ASYNCcnt-tracked). Guarded so the host
// pass / older toolchains compile the synchronous fallback.
#if __has_builtin(__builtin_amdgcn_global_load_async_to_lds_b128) && \
    __has_builtin(__builtin_amdgcn_s_wait_asynccnt)
#define ASYNC_LDS 1
#else
#define ASYNC_LDS 0
#endif

#if __has_builtin(__builtin_amdgcn_sched_group_barrier)
#define SCHED_DS_THEN_WMMA()                                    \
    do {                                                        \
        __builtin_amdgcn_sched_group_barrier(0x100, 18, 0);     \
        __builtin_amdgcn_sched_group_barrier(0x008,  8, 0);     \
    } while (0)
#else
#define SCHED_DS_THEN_WMMA() do {} while (0)
#endif

// builtin signature: (int4 global* src, int4 shared* dst, imm offset, imm cpol)
#define GPTR(p) ((__attribute__((address_space(1))) v4i*)(void*)(p))
#define LPTR(p) ((__attribute__((address_space(3))) v4i*)(void*)(p))

// ---------------------------------------------------------------------------
// bf16 WMMA GEMM:  C[M,N] = act( A[M,K](bf16) @ Bt[N,K]^T(bf16) + bias )
//   - A row-major [M,K] (lda), Bt row-major [N,K] (ldb)  => C = A @ Bt^T
//   - batched via grid.z with element strides sA/sB/sC
//   - ACT:  0 identity, 1 relu
//   - BCHK: 1 -> clamp A rows to M-1 (tiny GEMMs); stores guarded by gr<M
//   - CTR:  1 -> store C transposed per batch-of-1024 rows (builds v^T)
// 8 waves/block; wave tile 16x128 (8 accumulators). Triple-buffered LDS fed
// by GLOBAL_LOAD_ASYNC_TO_LDS_B128: tiles t+1,t+2 in flight during compute
// of tile t; steady-state s_wait_asynccnt(4) retires exactly one tile.
// ---------------------------------------------------------------------------
template<int ACT, int BCHK, int CTR, typename CT>
__global__ __launch_bounds__(256)
void gemm_bf16_wmma(const __bf16* __restrict__ A, const __bf16* __restrict__ Bt,
                    const float* __restrict__ bias, CT* __restrict__ C,
                    int M, int K,
                    int lda, int ldb, int ldc,
                    long long sA, long long sB, long long sC)
{
    __shared__ alignas(32) __bf16 As[3][BM * BK];   // [m][k]
    __shared__ alignas(32) __bf16 Bs[3][BN * BK];   // [n][k]

    const int bz = blockIdx.z;
    A  += (long long)bz * sA;
    Bt += (long long)bz * sB;
    if (!CTR) C += (long long)bz * sC;

    const int tid  = threadIdx.x;
    const int wave = tid >> 5;
    const int lane = tid & 31;
    const int half = lane >> 4;      // 0: lanes 0-15, 1: lanes 16-31
    const int l16  = lane & 15;

    const int m_blk = blockIdx.y * BM;
    const int n_blk = blockIdx.x * BN;

    v8f acc[8] = {};

    const int nk = K / BK;

    // issue one 128x32 A tile + 128x32 B tile into LDS buffer `buf`
    auto issue = [&](int t, int buf) {
        const int k0 = t * BK;
        #pragma unroll
        for (int p = 0; p < 2; ++p) {
            int idx = tid + p * 256;          // 0..511
            int r   = idx >> 2;               // 0..127
            int c   = (idx & 3) * 8;          // 0,8,16,24
            int gm  = m_blk + r;
            if (BCHK) gm = (gm < M) ? gm : (M - 1);   // clamp; never stored
            const __bf16* ga = A  + (long long)gm * lda + (k0 + c);
            const __bf16* gb = Bt + (long long)(n_blk + r) * ldb + (k0 + c);
#if ASYNC_LDS
            __builtin_amdgcn_global_load_async_to_lds_b128(
                GPTR(ga), LPTR(&As[buf][r * BK + c]), 0, 0);
            __builtin_amdgcn_global_load_async_to_lds_b128(
                GPTR(gb), LPTR(&Bs[buf][r * BK + c]), 0, 0);
#else
            *(uint4*)(&As[buf][r * BK + c]) = *(const uint4*)ga;
            *(uint4*)(&Bs[buf][r * BK + c]) = *(const uint4*)gb;
#endif
        }
    };

    // 18 ds_load_b128 (2 A + 16 B) then 8 WMMAs on LDS buffer `buf`
    auto compute = [&](int buf) {
        union { v16bf v; uint4 q[2]; } af;
        const int am = wave * 16 + l16;
        af.q[0] = *(const uint4*)(&As[buf][am * BK + half * 16]);
        af.q[1] = *(const uint4*)(&As[buf][am * BK + half * 16 + 8]);

        v16bf bf[8];
        #pragma unroll
        for (int nf = 0; nf < 8; ++nf)
            bf[nf] = *(const v16bf*)(&Bs[buf][(nf * 16 + l16) * BK + half * 16]);
        #pragma unroll
        for (int nf = 0; nf < 8; ++nf)
            acc[nf] = __builtin_amdgcn_wmma_f32_16x16x32_bf16(
                false, af.v, false, bf[nf], (short)0, acc[nf], false, false);
        SCHED_DS_THEN_WMMA();
    };

    issue(0, 0);
    if (nk > 1) issue(1, 1);

    int ibuf = 2;                        // next buffer to fill
    int cbuf = 0;                        // next buffer to consume
    for (int t = 0; t < nk - 1; ++t) {   // steady state: one tile retires
#if ASYNC_LDS
        __builtin_amdgcn_s_wait_asynccnt(4);
#endif
        __syncthreads();
        if (t + 2 < nk) {
            issue(t + 2, ibuf);
            ibuf = (ibuf == 2) ? 0 : (ibuf + 1);
        }
        compute(cbuf);
        cbuf = (cbuf == 2) ? 0 : (cbuf + 1);
    }
    // final tile
#if ASYNC_LDS
    __builtin_amdgcn_s_wait_asynccnt(0);
#endif
    __syncthreads();
    compute(cbuf);

    // epilogue: C/D layout -> lane col = l16, VGPR r -> row half*8+r
    #pragma unroll
    for (int nf = 0; nf < 8; ++nf) {
        const int gc = n_blk + nf * 16 + l16;
        const float bv = bias ? bias[gc] : 0.0f;
        #pragma unroll
        for (int r = 0; r < 8; ++r) {
            const int gr = m_blk + wave * 16 + half * 8 + r;
            if (BCHK && gr >= M) continue;
            float x = acc[nf][r] + bv;
            if (ACT) x = x > 0.0f ? x : 0.0f;
            if (CTR) {
                const int bb = gr >> 10;
                const int kk = gr & 1023;
                C[(long long)bb * sC + (long long)gc * ldc + kk] = (CT)x;
            } else {
                C[(long long)gr * ldc + gc] = (CT)x;
            }
        }
    }
}

// ---------------------------------------------------------------------------
// Row softmax over 1024 logits with diagonal + key mask, bf16 output
// ---------------------------------------------------------------------------
__global__ __launch_bounds__(256)
void softmax_mask_kernel(const float* __restrict__ logits,
                         __bf16* __restrict__ probs,
                         const unsigned char* __restrict__ s_mask)
{
    const int row = blockIdx.x;           // b*1024 + m
    const int b   = row >> 10;
    const int m   = row & 1023;
    const float*  L = logits + (long long)row * 1024;
    __bf16*       P = probs  + (long long)row * 1024;

    __shared__ float red[8];
    float x[4];
    float mx = -3.4e38f;
    #pragma unroll
    for (int i = 0; i < 4; ++i) {
        int n = threadIdx.x + i * 256;
        float v = L[n];
        bool dead = (n == m) || (s_mask[b * 1024 + n] == 0);
        v = dead ? -3.4e38f : v;
        x[i] = v;
        mx = fmaxf(mx, v);
    }
    #pragma unroll
    for (int off = 16; off > 0; off >>= 1)
        mx = fmaxf(mx, __shfl_xor(mx, off, 32));
    if ((threadIdx.x & 31) == 0) red[threadIdx.x >> 5] = mx;
    __syncthreads();
    float bmx = red[0];
    #pragma unroll
    for (int w = 1; w < 8; ++w) bmx = fmaxf(bmx, red[w]);
    __syncthreads();

    float s = 0.0f;
    #pragma unroll
    for (int i = 0; i < 4; ++i) { x[i] = __expf(x[i] - bmx); s += x[i]; }
    #pragma unroll
    for (int off = 16; off > 0; off >>= 1)
        s += __shfl_xor(s, off, 32);
    if ((threadIdx.x & 31) == 0) red[threadIdx.x >> 5] = s;
    __syncthreads();
    float bs = 0.0f;
    #pragma unroll
    for (int w = 0; w < 8; ++w) bs += red[w];
    const float inv = 1.0f / bs;
    #pragma unroll
    for (int i = 0; i < 4; ++i) {
        int n = threadIdx.x + i * 256;
        P[n] = (__bf16)(x[i] * inv);
    }
}

// ---------------------------------------------------------------------------
// Small utility kernels
// ---------------------------------------------------------------------------
__global__ void cvt_f32_bf16_kernel(const float* __restrict__ s,
                                    __bf16* __restrict__ d, long long n)
{
    long long i = (long long)blockIdx.x * 256 + threadIdx.x;
    if (i < n) d[i] = (__bf16)s[i];
}

__global__ void fill_zero_bf16_kernel(__bf16* __restrict__ d, long long n)
{
    long long i = (long long)blockIdx.x * 256 + threadIdx.x;
    if (i < n) d[i] = (__bf16)0.0f;
}

// d[n*lddst + k] = (bf16) s[k*N + n]   -- weight [K,N] -> [N,K] transpose+cvt
__global__ void cvt_transpose_kernel(const float* __restrict__ s,
                                     __bf16* __restrict__ d,
                                     int K, int N, int lddst)
{
    long long i = (long long)blockIdx.x * 256 + threadIdx.x;
    if (i >= (long long)K * N) return;
    int k = (int)(i / N);
    int n = (int)(i - (long long)k * N);
    d[(long long)n * lddst + k] = (__bf16)s[i];
}

// d[r*lddst + c] = (bf16) s[r*ldsrc + c]
__global__ void cvt_strided_kernel(const float* __restrict__ s,
                                   __bf16* __restrict__ d,
                                   long long rows, int cols, int ldsrc, int lddst)
{
    long long i = (long long)blockIdx.x * 256 + threadIdx.x;
    if (i >= rows * (long long)cols) return;
    long long r = i / cols;
    int c = (int)(i - r * cols);
    d[r * lddst + c] = (__bf16)s[r * ldsrc + c];
}

// d[r*lddst + c] = s[r*ldsrc + c]   (f32 -> f32, output concat of s1)
__global__ void copy_f32_strided_kernel(const float* __restrict__ s,
                                        float* __restrict__ d,
                                        long long rows, int cols, int ldsrc, int lddst)
{
    long long i = (long long)blockIdx.x * 256 + threadIdx.x;
    if (i >= rows * (long long)cols) return;
    long long r = i / cols;
    int c = (int)(i - r * cols);
    d[r * lddst + c] = s[r * ldsrc + c];
}

// rhs_scaled[m,h] = rhs[m,h] * gqv[m>>10, h]
__global__ void scale_rhs_kernel(const __bf16* __restrict__ rhs,
                                 const float* __restrict__ gqv,
                                 __bf16* __restrict__ out,
                                 long long total, int H)
{
    long long i = (long long)blockIdx.x * 256 + threadIdx.x;
    if (i >= total) return;
    long long m = i / H;
    int h = (int)(i - m * H);
    int b = (int)(m >> 10);
    float r = (float)rhs[i];
    out[i] = (__bf16)(r * gqv[b * H + h]);
}

// ---------------------------------------------------------------------------
// Host-side orchestration
// ---------------------------------------------------------------------------
static const int Bn    = 16;
static const int Nn    = 1024;
static const int Mrows = Bn * Nn;       // 16384
static const int DS1   = 768;
static const int DIN   = 896;
static const int HATT  = 512;
static const int DBASE = 768;

extern "C" void kernel_launch(void* const* d_in, const int* in_sizes, int n_in,
                              void* d_out, int out_size, void* d_ws, size_t ws_size,
                              hipStream_t stream)
{
    (void)in_sizes; (void)n_in; (void)out_size; (void)ws_size;

    const float* s1      = (const float*)d_in[0];
    const float* boxes   = (const float*)d_in[1];
    const float* q       = (const float*)d_in[2];
    const unsigned char* s_mask = (const unsigned char*)d_in[3];
    const float* gb_w1   = (const float*)d_in[4];
    const float* gb_b1   = (const float*)d_in[5];
    const float* gb_w2   = (const float*)d_in[6];
    const float* gb_b2   = (const float*)d_in[7];
    const float* gs1_w1  = (const float*)d_in[8];
    const float* gs1_b1  = (const float*)d_in[9];
    const float* gs1_w2  = (const float*)d_in[10];
    const float* gs1_b2  = (const float*)d_in[11];
    const float* gs2_w1  = (const float*)d_in[12];
    const float* gs2_b1  = (const float*)d_in[13];
    const float* gs2_w2  = (const float*)d_in[14];
    const float* gs2_b2  = (const float*)d_in[15];
    const float* gq_w1   = (const float*)d_in[16];
    const float* gq_b1   = (const float*)d_in[17];
    const float* gq_w2   = (const float*)d_in[18];
    const float* gq_b2   = (const float*)d_in[19];
    const float* gs3_w1  = (const float*)d_in[20];
    const float* gs3_b1  = (const float*)d_in[21];
    const float* gs3_w2  = (const float*)d_in[22];
    const float* gs3_b2  = (const float*)d_in[23];
    float* out = (float*)d_out;

    // ---- workspace carve-up (256B aligned slabs)
    char* wp = (char*)d_ws;
    auto alloc = [&](size_t bytes) -> void* {
        void* r = (void*)wp;
        wp += (bytes + 255) & ~(size_t)255;
        return r;
    };
    const long long M = Mrows;
    __bf16* boxesp  = (__bf16*)alloc((size_t)M * 32 * 2);        // [M,32] padded A
    __bf16* gbw1t   = (__bf16*)alloc((size_t)128 * 32 * 2);      // [128,32] padded B^T
    __bf16* gbw2t   = (__bf16*)alloc((size_t)128 * 128 * 2);
    __bf16* bsh     = (__bf16*)alloc((size_t)M * 128 * 2);
    __bf16* cat     = (__bf16*)alloc((size_t)M * DIN * 2);
    __bf16* w_gs1_1 = (__bf16*)alloc((size_t)HATT * DIN * 2);    // all weights [N,K]
    __bf16* w_gs1_2 = (__bf16*)alloc((size_t)HATT * HATT * 2);
    __bf16* w_gs2_1 = (__bf16*)alloc((size_t)HATT * DIN * 2);
    __bf16* w_gs2_2 = (__bf16*)alloc((size_t)HATT * HATT * 2);
    __bf16* w_gq_1  = (__bf16*)alloc((size_t)HATT * DS1 * 2);
    __bf16* w_gq_2  = (__bf16*)alloc((size_t)HATT * HATT * 2);
    __bf16* w_gs3_1 = (__bf16*)alloc((size_t)HATT * DS1 * 2);
    __bf16* w_gs3_2 = (__bf16*)alloc((size_t)DBASE * HATT * 2);
    __bf16* qbf     = (__bf16*)alloc((size_t)Bn * DS1 * 2);
    __bf16* gqh     = (__bf16*)alloc((size_t)Bn * HATT * 2);
    float*  gqvf    = (float*)alloc((size_t)Bn * HATT * 4);
    __bf16* hidden  = (__bf16*)alloc((size_t)M * HATT * 2);      // reused 3x
    __bf16* lhs     = (__bf16*)alloc((size_t)M * HATT * 2);
    __bf16* rhs     = (__bf16*)alloc((size_t)M * HATT * 2);
    __bf16* rhss    = (__bf16*)alloc((size_t)M * HATT * 2);
    __bf16* vT      = (__bf16*)alloc((size_t)Bn * DBASE * Nn * 2); // [B][768][1024]
    float*  logits  = (float*)alloc((size_t)Bn * Nn * Nn * 4);
    __bf16* probs   = (__bf16*)alloc((size_t)Bn * Nn * Nn * 2);

    auto cvt = [&](const float* s, __bf16* d, long long n) {
        cvt_f32_bf16_kernel<<<(unsigned)((n + 255) / 256), 256, 0, stream>>>(s, d, n);
    };
    auto cvtT = [&](const float* s, __bf16* d, int K, int N, int lddst) {
        long long n = (long long)K * N;
        cvt_transpose_kernel<<<(unsigned)((n + 255) / 256), 256, 0, stream>>>(s, d, K, N, lddst);
    };
    auto zero = [&](__bf16* d, long long n) {
        fill_zero_bf16_kernel<<<(unsigned)((n + 255) / 256), 256, 0, stream>>>(d, n);
    };

    // ---- weight conversion + transpose to [N,K]
    cvtT(gb_w2,  gbw2t,   128, 128, 128);
    cvtT(gs1_w1, w_gs1_1, DIN, HATT, DIN);
    cvtT(gs1_w2, w_gs1_2, HATT, HATT, HATT);
    cvtT(gs2_w1, w_gs2_1, DIN, HATT, DIN);
    cvtT(gs2_w2, w_gs2_2, HATT, HATT, HATT);
    cvtT(gq_w1,  w_gq_1,  DS1, HATT, DS1);
    cvtT(gq_w2,  w_gq_2,  HATT, HATT, HATT);
    cvtT(gs3_w1, w_gs3_1, DS1, HATT, DS1);
    cvtT(gs3_w2, w_gs3_2, HATT, DBASE, HATT);
    cvt(q, qbf, (long long)Bn * DS1);

    // ---- K-padded box operands
    zero(boxesp, M * 32);
    zero(gbw1t, 128 * 32);
    {
        long long n = M * 10;
        cvt_strided_kernel<<<(unsigned)((n + 255) / 256), 256, 0, stream>>>(
            boxes, boxesp, M, 10, 10, 32);
        cvtT(gb_w1, gbw1t, 10, 128, 32);          // [10,128] -> [128,32-pad]
        n = M * DS1;
        cvt_strided_kernel<<<(unsigned)((n + 255) / 256), 256, 0, stream>>>(
            s1, cat, M, DS1, DS1, DIN);           // cat[:, 0:768] = s1
    }

    auto ggrid = [&](int Mr, int Nc, int batches) {
        return dim3((unsigned)(Nc / BN), (unsigned)((Mr + BM - 1) / BM), (unsigned)batches);
    };

    // ---- box MLP -> cat[:, 768:896]
    gemm_bf16_wmma<1, 0, 0, __bf16><<<ggrid(Mrows, 128, 1), 256, 0, stream>>>(
        boxesp, gbw1t, gb_b1, bsh, Mrows, 32, 32, 32, 128, 0, 0, 0);
    gemm_bf16_wmma<0, 0, 0, __bf16><<<ggrid(Mrows, 128, 1), 256, 0, stream>>>(
        bsh, gbw2t, gb_b2, cat + DS1, Mrows, 128, 128, 128, DIN, 0, 0, 0);

    // ---- lhs = MLP_gs1(cat)
    gemm_bf16_wmma<1, 0, 0, __bf16><<<ggrid(Mrows, HATT, 1), 256, 0, stream>>>(
        cat, w_gs1_1, gs1_b1, hidden, Mrows, DIN, DIN, DIN, HATT, 0, 0, 0);
    gemm_bf16_wmma<0, 0, 0, __bf16><<<ggrid(Mrows, HATT, 1), 256, 0, stream>>>(
        hidden, w_gs1_2, gs1_b2, lhs, Mrows, HATT, HATT, HATT, HATT, 0, 0, 0);

    // ---- rhs = MLP_gs2(cat)
    gemm_bf16_wmma<1, 0, 0, __bf16><<<ggrid(Mrows, HATT, 1), 256, 0, stream>>>(
        cat, w_gs2_1, gs2_b1, hidden, Mrows, DIN, DIN, DIN, HATT, 0, 0, 0);
    gemm_bf16_wmma<0, 0, 0, __bf16><<<ggrid(Mrows, HATT, 1), 256, 0, stream>>>(
        hidden, w_gs2_2, gs2_b2, rhs, Mrows, HATT, HATT, HATT, HATT, 0, 0, 0);

    // ---- gqv = MLP_gq(q)  (M=16 -> clamped path; f32 output)
    gemm_bf16_wmma<1, 1, 0, __bf16><<<ggrid(Bn, HATT, 1), 256, 0, stream>>>(
        qbf, w_gq_1, gq_b1, gqh, Bn, DS1, DS1, DS1, HATT, 0, 0, 0);
    gemm_bf16_wmma<0, 1, 0, float><<<ggrid(Bn, HATT, 1), 256, 0, stream>>>(
        gqh, w_gq_2, gq_b2, gqvf, Bn, HATT, HATT, HATT, HATT, 0, 0, 0);

    // ---- rhss = rhs * gqv[batch]
    {
        long long n = M * HATT;
        scale_rhs_kernel<<<(unsigned)((n + 255) / 256), 256, 0, stream>>>(
            rhs, gqvf, rhss, n, HATT);
    }

    // ---- v^T = MLP_gs3(s1)^T  (s1 lives in cat[:, 0:768], lda=896)
    gemm_bf16_wmma<1, 0, 0, __bf16><<<ggrid(Mrows, HATT, 1), 256, 0, stream>>>(
        cat, w_gs3_1, gs3_b1, hidden, Mrows, DS1, DIN, DS1, HATT, 0, 0, 0);
    gemm_bf16_wmma<0, 0, 1, __bf16><<<ggrid(Mrows, DBASE, 1), 256, 0, stream>>>(
        hidden, w_gs3_2, gs3_b2, vT, Mrows, HATT, HATT, HATT, Nn,
        0, 0, (long long)DBASE * Nn);

    // ---- logits[b] = lhs[b] @ rhss[b]^T   (batched; B operand is [N,K])
    gemm_bf16_wmma<0, 0, 0, float><<<ggrid(Nn, Nn, Bn), 256, 0, stream>>>(
        lhs, rhss, nullptr, logits, Nn, HATT, HATT, HATT, Nn,
        (long long)Nn * HATT, (long long)Nn * HATT, (long long)Nn * Nn);

    // ---- masked softmax -> bf16 probs
    softmax_mask_kernel<<<(unsigned)(Bn * Nn), 256, 0, stream>>>(logits, probs, s_mask);

    // ---- out[:, 768:1536] = probs[b] @ v[b] = probs[b] @ (v^T[b])^T (batched)
    gemm_bf16_wmma<0, 0, 0, float><<<ggrid(Nn, DBASE, Bn), 256, 0, stream>>>(
        probs, vT, nullptr, out + DS1, Nn, Nn, Nn, Nn, DS1 + DBASE,
        (long long)Nn * Nn, (long long)DBASE * Nn, (long long)Nn * (DS1 + DBASE));

    // ---- out[:, 0:768] = s1
    {
        long long n = M * DS1;
        copy_f32_strided_kernel<<<(unsigned)((n + 255) / 256), 256, 0, stream>>>(
            s1, out, M, DS1, DS1, DS1 + DBASE);
    }
}
